// LRPModel_71846212928131
// MI455X (gfx1250) — matrix operand order repaired and checked by reading, not compiled, not run on hardware
//
#include <hip/hip_runtime.h>

// ---------------------------------------------------------------------------
// LRP head for MI455X (gfx1250, wave32), bf16 WMMA with f32 accumulation.
//   prep:    act -> bf16, W -> bf16 [C_PAD][D], w_augT -> bf16 [N_PAD][C_PAD]
//   GEMM1:   logits = act @ W^T + bias          (V_WMMA_F32_16X16X32_BF16)
//   softmax: S = softmax(logits)/logits         (f32 rowwise, bf16 out)
//   GEMM2:   core = S @ w_aug; rel = act_aug*core; emit act_aug + rel
// Block tile 128x128, 8 waves, each wave 32x64 (2x4 WMMA tiles), K-step 32.
// Double-buffered LDS: next tile's global_load_b128 issue before current
// tile's WMMA, so VMEM latency overlaps matrix math.
// Fragment gathers follow ISA 7.12.2 16-bit A(16x32)/B(32x16) VGPR layouts.
// ---------------------------------------------------------------------------

typedef __bf16  bf16_t;
typedef bf16_t  v16bf __attribute__((ext_vector_type(16)));
typedef float   v8f   __attribute__((ext_vector_type(8)));

#define B_ROWS 8192
#define D_DIM  2048
#define C_DIM  1000
#define C_PAD  1024   // K padding for GEMM2 / N padding for GEMM1 (zeros -> exact)
#define N_AUG  2049   // D+1 (bias column)
#define N_PAD  2176   // 17 * 128, N-tile padding for GEMM2
#define LDS_STRIDE 34 // 32 + 2 bf16 pad

__device__ __forceinline__ unsigned short f2bf(float x) {
  unsigned int u = __float_as_uint(x);
  u += 0x7FFFu + ((u >> 16) & 1u);         // round-to-nearest-even
  return (unsigned short)(u >> 16);
}

union FragU { unsigned int w[8]; v16bf v; };

// A-matrix 16x32 bf16 fragment (ISA 7.12.2): lane = half*16 + m,
// VGPR v holds K pair at ((v>=4)*16 + half*8 + (v%4)*2).
__device__ __forceinline__ v16bf frag_a(const unsigned short* tile, int m0, int lane) {
  const int half = lane >> 4, mr = lane & 15;
  const unsigned short* rowp = tile + (m0 + mr) * LDS_STRIDE;
  FragU f;
#pragma unroll
  for (int v = 0; v < 8; ++v) {
    const int kk = ((v >> 2) << 4) + (half << 3) + ((v & 3) << 1);
    f.w[v] = *(const unsigned int*)(rowp + kk);
  }
  return f.v;
}

// B-matrix 32x16 bf16 fragment: lane = half*16 + n,
// VGPR v holds K pair at ((v>=4)*16 + half*8 + (v%4)*2).
__device__ __forceinline__ v16bf frag_b(const unsigned short* tile, int n0, int lane) {
  const int half = lane >> 4, nc = lane & 15;
  const unsigned short* rowp = tile + (n0 + nc) * LDS_STRIDE;
  FragU f;
#pragma unroll
  for (int v = 0; v < 8; ++v) {
    const int kk = ((v >> 2) << 4) + (half << 3) + ((v & 3) << 1);
    f.w[v] = *(const unsigned int*)(rowp + kk);
  }
  return f.v;
}

#define WMMA_BF16(A, Bm, Cacc) \
  __builtin_amdgcn_wmma_f32_16x16x32_bf16(false, (A), false, (Bm), (short)0, (Cacc), false, false)

// Per-thread register staging for one 128x32 A-tile + 128x32 B-tile.
struct TileRegs { uint4 a[2]; uint4 b[2]; };

__device__ __forceinline__ void load_tiles(TileRegs& t,
                                           const unsigned short* __restrict__ Asrc, size_t strideA,
                                           const unsigned short* __restrict__ Bsrc, size_t strideB,
                                           int mBase, int nBase, int kt, int lrow, int lcv) {
#pragma unroll
  for (int i = 0; i < 2; ++i) {
    const int row = lrow + i * 64;
    t.a[i] = *(const uint4*)(Asrc + (size_t)(mBase + row) * strideA + kt + lcv);
    t.b[i] = *(const uint4*)(Bsrc + (size_t)(nBase + row) * strideB + kt + lcv);
  }
}

__device__ __forceinline__ void store_tiles(unsigned short* As, unsigned short* Bs,
                                            const TileRegs& t, int lrow, int lcv) {
#pragma unroll
  for (int i = 0; i < 2; ++i) {
    const int row = lrow + i * 64;
    unsigned short* da = &As[row * LDS_STRIDE + lcv];
    *(unsigned int*)(da + 0) = t.a[i].x;
    *(unsigned int*)(da + 2) = t.a[i].y;
    *(unsigned int*)(da + 4) = t.a[i].z;
    *(unsigned int*)(da + 6) = t.a[i].w;
    unsigned short* db = &Bs[row * LDS_STRIDE + lcv];
    *(unsigned int*)(db + 0) = t.b[i].x;
    *(unsigned int*)(db + 2) = t.b[i].y;
    *(unsigned int*)(db + 4) = t.b[i].z;
    *(unsigned int*)(db + 6) = t.b[i].w;
  }
}

__device__ __forceinline__ void mma_step(const unsigned short* As, const unsigned short* Bs,
                                         int wm, int wn, int lane, v8f acc[2][4]) {
  const v16bf a0 = frag_a(As, wm, lane);
  const v16bf a1 = frag_a(As, wm + 16, lane);
#pragma unroll
  for (int j = 0; j < 4; ++j) {
    const v16bf bj = frag_b(Bs, wn + j * 16, lane);
    acc[0][j] = WMMA_BF16(a0, bj, acc[0][j]);
    acc[1][j] = WMMA_BF16(a1, bj, acc[1][j]);
  }
}

// ---------------------------------------------------------------------------
// Prep 1: act (f32 [B][D]) -> act_bf16, 4 elements per thread.
// ---------------------------------------------------------------------------
__global__ __launch_bounds__(256) void lrp_prep_act(const float* __restrict__ act,
                                                    unsigned short* __restrict__ actbf) {
  const size_t base = ((size_t)blockIdx.x * 256 + threadIdx.x) * 4;
  const float4 f = *(const float4*)(act + base);
  uint2 p;
  p.x = ((unsigned int)f2bf(f.y) << 16) | f2bf(f.x);
  p.y = ((unsigned int)f2bf(f.w) << 16) | f2bf(f.z);
  *(uint2*)(actbf + base) = p;
}

// ---------------------------------------------------------------------------
// Prep 2: W (f32 [C][D]) -> wbf [C_PAD][D] bf16, rows >= C_DIM zero.
// ---------------------------------------------------------------------------
__global__ __launch_bounds__(256) void lrp_prep_w(const float* __restrict__ weight,
                                                  unsigned short* __restrict__ wbf) {
  const size_t base = ((size_t)blockIdx.x * 256 + threadIdx.x) * 4;  // [0, C_PAD*D)
  const int c = (int)(base >> 11);
  float4 f = make_float4(0.f, 0.f, 0.f, 0.f);
  if (c < C_DIM) f = *(const float4*)(weight + base);
  uint2 p;
  p.x = ((unsigned int)f2bf(f.y) << 16) | f2bf(f.x);
  p.y = ((unsigned int)f2bf(f.w) << 16) | f2bf(f.z);
  *(uint2*)(wbf + base) = p;
}

// ---------------------------------------------------------------------------
// Prep 3: w_augT[d][c] (bf16 [N_PAD][C_PAD]) = weight[c][d] / bias[c] / 0 pad.
// ---------------------------------------------------------------------------
__global__ __launch_bounds__(256) void lrp_prep_waugT(const float* __restrict__ weight,
                                                      const float* __restrict__ bias,
                                                      unsigned short* __restrict__ waugT) {
  const int idx = blockIdx.x * 256 + threadIdx.x;   // [0, N_PAD*C_PAD)
  const int d = idx >> 10;
  const int c = idx & (C_PAD - 1);
  float v = 0.f;
  if (c < C_DIM) {
    if (d < D_DIM)       v = weight[(size_t)c * D_DIM + d];
    else if (d == D_DIM) v = bias[c];
  }
  waugT[idx] = f2bf(v);
}

// ---------------------------------------------------------------------------
// GEMM1: logits[b][c] = sum_d act[b][d]*W[c][d] + bias[c] -> f32 ws [B][C_PAD]
// A = actbf [B][D], B = wbf [C_PAD][D] (row-major = contiguous K per column).
// ---------------------------------------------------------------------------
__global__ __launch_bounds__(256) void lrp_gemm1(const unsigned short* __restrict__ actbf,
                                                 const unsigned short* __restrict__ wbf,
                                                 const float* __restrict__ bias,
                                                 float* __restrict__ logits) {
  __shared__ unsigned short As[2][128 * LDS_STRIDE];
  __shared__ unsigned short Bs[2][128 * LDS_STRIDE];

  const int tid  = threadIdx.x;
  const int lane = tid & 31;
  const int wid  = tid >> 5;
  const int wm   = (wid & 3) * 32;   // wave M offset
  const int wn   = (wid >> 2) * 64;  // wave N offset
  const int mBase = blockIdx.y * 128;
  const int nBase = blockIdx.x * 128;
  const int lrow = tid >> 2, lcv = (tid & 3) * 8;

  const v8f vzero = {0.f, 0.f, 0.f, 0.f, 0.f, 0.f, 0.f, 0.f};
  v8f acc[2][4];
#pragma unroll
  for (int i = 0; i < 2; ++i)
#pragma unroll
    for (int j = 0; j < 4; ++j) acc[i][j] = vzero;

  TileRegs t;
  load_tiles(t, actbf, D_DIM, wbf, D_DIM, mBase, nBase, 0, lrow, lcv);
  store_tiles(As[0], Bs[0], t, lrow, lcv);
  __syncthreads();

  int cur = 0;
  for (int kt = 0; kt < D_DIM; kt += 32) {
    const bool has_next = (kt + 32 < D_DIM);
    if (has_next) {
      load_tiles(t, actbf, D_DIM, wbf, D_DIM, mBase, nBase, kt + 32, lrow, lcv);
      __builtin_prefetch(actbf + (size_t)(mBase + lrow) * D_DIM + kt + 64, 0, 1);
    }
    mma_step(As[cur], Bs[cur], wm, wn, lane, acc);
    if (has_next) {
      __syncthreads();                       // everyone done reading buf[cur^1]
      store_tiles(As[cur ^ 1], Bs[cur ^ 1], t, lrow, lcv);
      __syncthreads();
      cur ^= 1;
    }
  }

  const int half = lane >> 4;
#pragma unroll
  for (int j = 0; j < 4; ++j) {
    const int ng = nBase + wn + j * 16 + (lane & 15);
    const float bv = (ng < C_DIM) ? bias[ng] : 0.f;
#pragma unroll
    for (int i = 0; i < 2; ++i) {
      const int mgb = mBase + wm + i * 16 + 8 * half;
#pragma unroll
      for (int r = 0; r < 8; ++r)
        logits[(size_t)(mgb + r) * C_PAD + ng] = acc[i][j][r] + bv;
    }
  }
}

// ---------------------------------------------------------------------------
// Softmax / divide: S[b][c] = softmax(logits[b])[c] / logits[b][c], bf16,
// zero-padded to C_PAD. One block per row.
// ---------------------------------------------------------------------------
__global__ __launch_bounds__(256) void lrp_softmax_div(const float* __restrict__ logits,
                                                       unsigned short* __restrict__ Sbf) {
  __shared__ float red[256];
  const int b = blockIdx.x, tid = threadIdx.x;
  const float NEG_INF = -__builtin_inff();

  float l[4];
#pragma unroll
  for (int i = 0; i < 4; ++i) {
    const int c = tid + i * 256;
    l[i] = (c < C_DIM) ? logits[(size_t)b * C_PAD + c] : NEG_INF;
  }
  float m = fmaxf(fmaxf(l[0], l[1]), fmaxf(l[2], l[3]));
  red[tid] = m;
  __syncthreads();
  for (int s = 128; s > 0; s >>= 1) {
    if (tid < s) red[tid] = fmaxf(red[tid], red[tid + s]);
    __syncthreads();
  }
  const float rowmax = red[0];
  __syncthreads();

  float e[4], sum = 0.f;
#pragma unroll
  for (int i = 0; i < 4; ++i) {
    e[i] = (l[i] == NEG_INF) ? 0.f : __expf(l[i] - rowmax);
    sum += e[i];
  }
  red[tid] = sum;
  __syncthreads();
  for (int s = 128; s > 0; s >>= 1) {
    if (tid < s) red[tid] += red[tid + s];
    __syncthreads();
  }
  const float inv = 1.f / red[0];

#pragma unroll
  for (int i = 0; i < 4; ++i) {
    const int c = tid + i * 256;
    float S = 0.f;
    if (c < C_DIM) S = (e[i] * inv) / l[i];   // probs / z_d  (eps = 0)
    Sbf[(size_t)b * C_PAD + c] = f2bf(S);
  }
}

// ---------------------------------------------------------------------------
// GEMM2 + epilogue: core[b][n] = sum_c S[b][c]*w_augT[n][c];
// rel = act_aug * core; also writes act_aug. M=B_ROWS, K=C_PAD, N=N_PAD.
// ---------------------------------------------------------------------------
__global__ __launch_bounds__(256) void lrp_gemm2(const unsigned short* __restrict__ Sbf,
                                                 const unsigned short* __restrict__ waugT,
                                                 const float* __restrict__ act,
                                                 float* __restrict__ out_act,
                                                 float* __restrict__ out_rel) {
  __shared__ unsigned short As[2][128 * LDS_STRIDE];
  __shared__ unsigned short Bs[2][128 * LDS_STRIDE];

  const int tid  = threadIdx.x;
  const int lane = tid & 31;
  const int wid  = tid >> 5;
  const int wm   = (wid & 3) * 32;
  const int wn   = (wid >> 2) * 64;
  const int mBase = blockIdx.y * 128;
  const int nBase = blockIdx.x * 128;
  const int lrow = tid >> 2, lcv = (tid & 3) * 8;

  const v8f vzero = {0.f, 0.f, 0.f, 0.f, 0.f, 0.f, 0.f, 0.f};
  v8f acc[2][4];
#pragma unroll
  for (int i = 0; i < 2; ++i)
#pragma unroll
    for (int j = 0; j < 4; ++j) acc[i][j] = vzero;

  TileRegs t;
  load_tiles(t, Sbf, C_PAD, waugT, C_PAD, mBase, nBase, 0, lrow, lcv);
  store_tiles(As[0], Bs[0], t, lrow, lcv);
  __syncthreads();

  int cur = 0;
  for (int kt = 0; kt < C_PAD; kt += 32) {
    const bool has_next = (kt + 32 < C_PAD);
    if (has_next) {
      load_tiles(t, Sbf, C_PAD, waugT, C_PAD, mBase, nBase, kt + 32, lrow, lcv);
      __builtin_prefetch(Sbf + (size_t)(mBase + lrow) * C_PAD + kt + 64, 0, 1);
    }
    mma_step(As[cur], Bs[cur], wm, wn, lane, acc);
    if (has_next) {
      __syncthreads();
      store_tiles(As[cur ^ 1], Bs[cur ^ 1], t, lrow, lcv);
      __syncthreads();
      cur ^= 1;
    }
  }

  const int half = lane >> 4;
#pragma unroll
  for (int j = 0; j < 4; ++j) {
    const int ng = nBase + wn + j * 16 + (lane & 15);
    if (ng < N_AUG) {
#pragma unroll
      for (int i = 0; i < 2; ++i) {
        const int mgb = mBase + wm + i * 16 + 8 * half;
#pragma unroll
        for (int r = 0; r < 8; ++r) {
          const int mg = mgb + r;
          const float aval = (ng < D_DIM) ? act[(size_t)mg * D_DIM + ng] : 1.0f;
          const size_t o = (size_t)mg * N_AUG + ng;
          out_act[o] = aval;
          out_rel[o] = aval * acc[i][j][r];
        }
      }
    }
  }
}

// ---------------------------------------------------------------------------
// Launch. ws layout (bytes):
//   [0,          33554432)   act_bf16 [8192][2048]
//   [33554432,   37748736)   W_bf16   [1024][2048]
//   [37748736,   71303168)   logits f32 [8192][1024]
//   [71303168,   88080384)   S bf16   [8192][1024]
//   [88080384,   92536832)   w_augT bf16 [2176][1024]
// ---------------------------------------------------------------------------
extern "C" void kernel_launch(void* const* d_in, const int* in_sizes, int n_in,
                              void* d_out, int out_size, void* d_ws, size_t ws_size,
                              hipStream_t stream) {
  (void)in_sizes; (void)n_in; (void)out_size; (void)ws_size;
  const float* act    = (const float*)d_in[0];   // [8192][2048]
  const float* weight = (const float*)d_in[1];   // [1000][2048]
  const float* bias   = (const float*)d_in[2];   // [1000]

  float* out_act = (float*)d_out;                          // act_aug [8192][2049]
  float* out_rel = out_act + (size_t)B_ROWS * N_AUG;       // rel     [8192][2049]

  char* ws = (char*)d_ws;
  unsigned short* actbf  = (unsigned short*)(ws);
  unsigned short* wbf    = (unsigned short*)(ws + 33554432u);
  float*          logits = (float*)         (ws + 37748736u);
  unsigned short* Sbf    = (unsigned short*)(ws + 71303168u);
  unsigned short* waugT  = (unsigned short*)(ws + 88080384u);

  lrp_prep_act  <<<(B_ROWS * D_DIM / 4) / 256, 256, 0, stream>>>(act, actbf);
  lrp_prep_w    <<<(C_PAD * D_DIM / 4) / 256, 256, 0, stream>>>(weight, wbf);
  lrp_prep_waugT<<<(N_PAD * C_PAD) / 256, 256, 0, stream>>>(weight, bias, waugT);
  lrp_gemm1     <<<dim3(C_PAD / 128, B_ROWS / 128), 256, 0, stream>>>(actbf, wbf, bias, logits);
  lrp_softmax_div<<<B_ROWS, 256, 0, stream>>>(logits, Sbf);
  lrp_gemm2     <<<dim3(N_PAD / 128, B_ROWS / 128), 256, 0, stream>>>(Sbf, waugT, act,
                                                                      out_act, out_rel);
}
